// GCNClassifier_29549374997129
// MI455X (gfx1250) — matrix operand order repaired and checked by reading, not compiled, not run on hardware
//
#include <hip/hip_runtime.h>
#include <hip/hip_bf16.h>

typedef __attribute__((ext_vector_type(16))) _Float16 v16h;
typedef __attribute__((ext_vector_type(8)))  float    v8f;

#define DFEAT      128
#define NEG_SLOPE  0.01f
#define LDSW       144   // padded LDS stride (halves); 288B per column -> 16B aligned

// ---------------------------------------------------------------------------
// H = act(X) @ W   (M x 128) x (128 x 128), fp32 in/out, f16 WMMA compute.
// One wave -> one 16-row x 128-col output tile. 8 waves / block.
// ---------------------------------------------------------------------------
__global__ __launch_bounds__(256)
void gemm_node_kernel(const float* __restrict__ X, const float* __restrict__ W,
                      float* __restrict__ H, int nrows, int apply_leaky)
{
    __shared__ _Float16 lw[DFEAT * LDSW];   // W transposed: lw[n*LDSW + k] = W[k][n]

    for (int i = threadIdx.x; i < DFEAT * DFEAT; i += 256) {
        int n = i >> 7, k = i & 127;
        lw[n * LDSW + k] = (_Float16)W[k * DFEAT + n];
    }
    __syncthreads();

    const int wave = threadIdx.x >> 5;
    const int lane = threadIdx.x & 31;
    const int tile = blockIdx.x * 8 + wave;           // 16-row tile index
    const int row0 = tile * 16;
    if (row0 >= nrows) return;

    const int hl = lane & 15;      // row (A) / col (B,D) within tile
    const int hi = lane >> 4;      // half-wave select

    v8f acc[8] = {};

    #pragma unroll
    for (int kt = 0; kt < 4; ++kt) {
        // ---- A fragment: 16x32 f16, per ISA layout ----
        v16h a;
        const float* arow = X + (size_t)(row0 + hl) * DFEAT + kt * 32 + hi * 8;
        #pragma unroll
        for (int i = 0; i < 8; ++i) {
            float v0 = arow[i];
            float v1 = arow[16 + i];
            if (apply_leaky) {
                v0 = (v0 >= 0.0f) ? v0 : NEG_SLOPE * v0;
                v1 = (v1 >= 0.0f) ? v1 : NEG_SLOPE * v1;
            }
            a[i]     = (_Float16)v0;   // K = kt*32 + hi*8 + i
            a[8 + i] = (_Float16)v1;   // K = kt*32 + hi*8 + 16 + i
        }
        #pragma unroll
        for (int nt = 0; nt < 8; ++nt) {
            // ---- B fragment: 32x16 f16, column nt*16+hl, K = kt*32 + hi*8 + i (+16 for i>=8) ----
            const _Float16* bp = lw + (nt * 16 + hl) * LDSW + kt * 32 + hi * 8;
            v16h b;
            #pragma unroll
            for (int i = 0; i < 8; ++i) { b[i] = bp[i]; b[8 + i] = bp[16 + i]; }
            acc[nt] = __builtin_amdgcn_wmma_f32_16x16x32_f16(
                false, a, false, b, (short)0, acc[nt], false, false);
        }
    }

    // ---- Store D: VGPR i -> row (row0 + hi*8 + i), col (nt*16 + hl) ----
    #pragma unroll
    for (int nt = 0; nt < 8; ++nt) {
        const int col = nt * 16 + hl;
        float* hrow = H + (size_t)(row0 + hi * 8) * DFEAT + col;
        #pragma unroll
        for (int i = 0; i < 8; ++i) hrow[(size_t)i * DFEAT] = acc[nt][i];
    }
}

// ---------------------------------------------------------------------------
// Degree accumulation (in-degree of dst), then dinv = 1/(deg+1), dsq = rsqrt(deg+1)
// ---------------------------------------------------------------------------
__global__ void deg_accum_kernel(const int* __restrict__ dst, float* __restrict__ deg, int nedges)
{
    int e = blockIdx.x * blockDim.x + threadIdx.x;
    if (e < nedges) atomicAdd(&deg[dst[e]], 1.0f);
}

__global__ void deg_finalize_kernel(float* __restrict__ dinv, float* __restrict__ dsq, int n)
{
    int i = blockIdx.x * blockDim.x + threadIdx.x;
    if (i < n) {
        float d = dinv[i] + 1.0f;      // + self loop
        dinv[i] = 1.0f / d;            // in place
        dsq[i]  = rsqrtf(d);
    }
}

// ---------------------------------------------------------------------------
// AGG = H * dinv[node] + b[feat]   (self-loop + bias init; no zeroing needed)
// ---------------------------------------------------------------------------
__global__ void selfloop_bias_kernel(const float* __restrict__ H, const float* __restrict__ dinv,
                                     const float* __restrict__ b, float* __restrict__ AGG,
                                     long total)
{
    long idx = (long)blockIdx.x * blockDim.x + threadIdx.x;
    if (idx < total) {
        int node = (int)(idx >> 7), f = (int)(idx & 127);
        AGG[idx] = H[idx] * dinv[node] + b[f];
    }
}

// ---------------------------------------------------------------------------
// Edge scatter: AGG[dst] += H[src] * dsq[src]*dsq[dst].  One wave per edge,
// 32 lanes x float4 = 128 floats.  Gathers/atomics are L2-resident (20MB rows).
// ---------------------------------------------------------------------------
__global__ __launch_bounds__(256)
void scatter_kernel(const float* __restrict__ H, float* __restrict__ AGG,
                    const int* __restrict__ src, const int* __restrict__ dst,
                    const float* __restrict__ dsq, int nedges)
{
    int edge = blockIdx.x * 8 + (threadIdx.x >> 5);
    if (edge >= nedges) return;
    int lane = threadIdx.x & 31;
    int s = src[edge];
    int d = dst[edge];
    float norm = dsq[s] * dsq[d];
    const float4 v = ((const float4*)(H + (size_t)s * DFEAT))[lane];
    float* ad = AGG + (size_t)d * DFEAT + lane * 4;
    atomicAdd(&ad[0], v.x * norm);
    atomicAdd(&ad[1], v.y * norm);
    atomicAdd(&ad[2], v.z * norm);
    atomicAdd(&ad[3], v.w * norm);
}

// ---------------------------------------------------------------------------
// Pooling: sums[batch[n]] += A[n], counts[batch[n]] += 1
// ---------------------------------------------------------------------------
__global__ __launch_bounds__(128)
void pool_kernel(const float* __restrict__ A, const int* __restrict__ batch,
                 float* __restrict__ sums, float* __restrict__ counts, int nnodes)
{
    int node = blockIdx.x;
    if (node >= nnodes) return;
    int f = threadIdx.x;
    int g = batch[node];
    atomicAdd(&sums[(size_t)g * DFEAT + f], A[(size_t)node * DFEAT + f]);
    if (f == 0) atomicAdd(&counts[g], 1.0f);
}

// embeds[g] = concat(sums[g], sums[g]/max(counts[g],1))   -> [G, 256]
__global__ void embed_kernel(const float* __restrict__ sums, const float* __restrict__ counts,
                             float* __restrict__ embeds, int ngraphs)
{
    int idx = blockIdx.x * blockDim.x + threadIdx.x;
    if (idx >= ngraphs * 256) return;
    int g = idx >> 8, f = idx & 255;
    float c = counts[g];
    c = (c < 1.0f) ? 1.0f : c;
    float sv = sums[(size_t)g * DFEAT + (f & 127)];
    embeds[idx] = (f < DFEAT) ? sv : sv / c;
}

// z = relu(embeds @ Wlin + blin)   [G,256] x [256,128]
__global__ void mlp1_kernel(const float* __restrict__ embeds, const float* __restrict__ Wlin,
                            const float* __restrict__ blin, float* __restrict__ z, int ngraphs)
{
    int idx = blockIdx.x * blockDim.x + threadIdx.x;
    if (idx >= ngraphs * DFEAT) return;
    int g = idx >> 7, j = idx & 127;
    const float* e = embeds + (size_t)g * 256;
    float acc = blin[j];
    #pragma unroll 8
    for (int k = 0; k < 256; ++k) acc += e[k] * Wlin[(size_t)k * DFEAT + j];
    z[idx] = (acc > 0.0f) ? acc : 0.0f;
}

// logits = z @ Wout + bout ; probs = softmax(logits).  One thread per graph.
__global__ void mlp2_kernel(const float* __restrict__ z, const float* __restrict__ Wout,
                            const float* __restrict__ bout, float* __restrict__ logits,
                            float* __restrict__ probs, int ngraphs, int nclasses)
{
    int g = blockIdx.x * blockDim.x + threadIdx.x;
    if (g >= ngraphs) return;
    const float* zr = z + (size_t)g * DFEAT;
    float l[16];
    float m = -3.0e38f;
    for (int c = 0; c < nclasses; ++c) {
        float acc = bout[c];
        #pragma unroll 8
        for (int k = 0; k < DFEAT; ++k) acc += zr[k] * Wout[(size_t)k * nclasses + c];
        l[c] = acc;
        m = fmaxf(m, acc);
    }
    float s = 0.0f;
    for (int c = 0; c < nclasses; ++c) { l[c] = __expf(l[c] - m); s += l[c]; }
    float inv = 1.0f / s;
    for (int c = 0; c < nclasses; ++c) {
        probs[(size_t)g * nclasses + c] = l[c] * inv;
    }
    // recompute exact logits for output (l[] was overwritten by exp)
    for (int c = 0; c < nclasses; ++c) {
        float acc = bout[c];
        #pragma unroll 8
        for (int k = 0; k < DFEAT; ++k) acc += zr[k] * Wout[(size_t)k * nclasses + c];
        logits[(size_t)g * nclasses + c] = acc;
    }
}

// ---------------------------------------------------------------------------
extern "C" void kernel_launch(void* const* d_in, const int* in_sizes, int n_in,
                              void* d_out, int out_size, void* d_ws, size_t ws_size,
                              hipStream_t stream)
{
    const float* x     = (const float*)d_in[0];
    const int*   edge  = (const int*)  d_in[1];   // [2, E] flat: src then dst
    const int*   batch = (const int*)  d_in[2];
    const float* W1    = (const float*)d_in[3];
    const float* b1    = (const float*)d_in[4];
    const float* W2    = (const float*)d_in[5];
    const float* b2    = (const float*)d_in[6];
    const float* W3    = (const float*)d_in[7];
    const float* b3    = (const float*)d_in[8];
    const float* Wlin  = (const float*)d_in[9];
    const float* blin  = (const float*)d_in[10];
    const float* Wout  = (const float*)d_in[11];
    const float* bout  = (const float*)d_in[12];

    const int E = in_sizes[1] / 2;
    const int N = in_sizes[2];
    const int G = 256, C = 10;
    const int* src = edge;
    const int* dst = edge + E;

    // workspace layout (floats)
    float*  w    = (float*)d_ws;
    size_t  NF   = (size_t)N * DFEAT;
    float*  B0   = w;                 // agg / activation buffer
    float*  B1   = B0 + NF;           // GEMM output H
    float*  dinv = B1 + NF;           // degree accum -> 1/deg (in place)
    float*  dsq  = dinv + N;          // rsqrt(deg)
    float*  sums = dsq + N;           // [G,128]
    float*  cnts = sums + (size_t)G * DFEAT;
    float*  z    = cnts + G;          // [G,128]

    float* out    = (float*)d_out;
    float* logits = out;              // [G,10]
    float* probs  = out + G * C;      // [G,10]
    float* embeds = out + 2 * G * C;  // [G,256]

    const int tiles  = (N + 15) / 16;
    const int gblk   = (tiles + 7) / 8;
    const int eblk   = (E + 7) / 8;          // 8 edge-waves per 256-thread block
    const long NFl   = (long)NF;
    const int  nfblk = (int)((NFl + 255) / 256);

    // degrees
    hipMemsetAsync(dinv, 0, (size_t)N * sizeof(float), stream);
    deg_accum_kernel<<<(E + 255) / 256, 256, 0, stream>>>(dst, dinv, E);
    deg_finalize_kernel<<<(N + 255) / 256, 256, 0, stream>>>(dinv, dsq, N);

    // layer 1 (input x, no input activation)
    gemm_node_kernel<<<gblk, 256, 0, stream>>>(x, W1, B1, N, 0);
    selfloop_bias_kernel<<<nfblk, 256, 0, stream>>>(B1, dinv, b1, B0, NFl);
    scatter_kernel<<<eblk, 256, 0, stream>>>(B1, B0, src, dst, dsq, E);

    // layer 2 (leaky applied to input inside GEMM load)
    gemm_node_kernel<<<gblk, 256, 0, stream>>>(B0, W2, B1, N, 1);
    selfloop_bias_kernel<<<nfblk, 256, 0, stream>>>(B1, dinv, b2, B0, NFl);
    scatter_kernel<<<eblk, 256, 0, stream>>>(B1, B0, src, dst, dsq, E);

    // layer 3
    gemm_node_kernel<<<gblk, 256, 0, stream>>>(B0, W3, B1, N, 1);
    selfloop_bias_kernel<<<nfblk, 256, 0, stream>>>(B1, dinv, b3, B0, NFl);
    scatter_kernel<<<eblk, 256, 0, stream>>>(B1, B0, src, dst, dsq, E);

    // pooling + embeds
    hipMemsetAsync(sums, 0, ((size_t)G * DFEAT + G) * sizeof(float), stream);
    pool_kernel<<<N, 128, 0, stream>>>(B0, batch, sums, cnts, N);
    embed_kernel<<<(G * 256 + 255) / 256, 256, 0, stream>>>(sums, cnts, embeds, G);

    // MLP head + softmax
    mlp1_kernel<<<(G * DFEAT + 255) / 256, 256, 0, stream>>>(embeds, Wlin, blin, z, G);
    mlp2_kernel<<<(G + 63) / 64, 64, 0, stream>>>(z, Wout, bout, logits, probs, G, C);
}
